// GraphConvolutionScoreNet_31061203485078
// MI455X (gfx1250) — compile-verified, matrix operand order checked by reading, not run on hardware
//
#include <hip/hip_runtime.h>
#include <hip/hip_bf16.h>

// Problem constants (match reference)
#define N_NODES 102400
#define N_EDGES 1638400
#define NGAUSS 50

typedef __attribute__((ext_vector_type(16))) _Float16 v16h;
typedef __attribute__((ext_vector_type(8)))  float    v8f;

__device__ __forceinline__ float softplusf(float v) {
  return (v > 20.0f) ? v : log1pf(__expf(v));
}
// order-preserving float<->uint for atomic max over signed floats
__device__ __forceinline__ unsigned fkey(float f) {
  unsigned u = __float_as_uint(f);
  return (u & 0x80000000u) ? ~u : (u | 0x80000000u);
}
__device__ __forceinline__ float finv(unsigned k) {
  return (k & 0x80000000u) ? __uint_as_float(k ^ 0x80000000u)
                           : __uint_as_float(~k);
}

// ---------------------------------------------------------------------------
// Pre-swizzle we[50,32] (f32, row-major) into the exact per-lane WMMA B
// operand layout, f16:  [layer][n_tile][k_step][lane][16 halves]
// so the hot kernel loads each B fragment as one contiguous 32B chunk.
// Half j of lane l (hi = l>>4, c = l&15) holds
//   B[k][n]  with  k = kstep*32 + hi*8 + (j&7) + ((j>>3)<<4),  n = nt*16 + c
__global__ void k_prep_bswz(const float* __restrict__ we1,
                            const float* __restrict__ we2,
                            _Float16* __restrict__ bswz) {
  int t = blockIdx.x * blockDim.x + threadIdx.x;
  if (t >= 2 * 2 * 2 * 32 * 16) return;
  int j = t & 15;
  int lane = (t >> 4) & 31;
  int kstep = (t >> 9) & 1;
  int nt = (t >> 10) & 1;
  int layer = t >> 11;
  int hi = lane >> 4, c = lane & 15;
  int k = kstep * 32 + hi * 8 + (j & 7) + ((j >> 3) << 4);
  int n = nt * 16 + c;
  const float* we = layer ? we2 : we1;
  float v = (k < NGAUSS) ? we[k * 32 + n] : 0.0f;
  bswz[t] = (_Float16)v;
}

// dist[e] = |pos[src]-pos[dst]|, deg[dst] += 1 (deg shared by all layers)
__global__ void k_dist_deg(const float* __restrict__ pos,
                           const int* __restrict__ ei,
                           float* __restrict__ dist, float* __restrict__ deg) {
  int e = blockIdx.x * blockDim.x + threadIdx.x;
  if (e >= N_EDGES) return;
  int s = ei[e], d = ei[N_EDGES + e];
  float dx = pos[s * 3 + 0] - pos[d * 3 + 0];
  float dy = pos[s * 3 + 1] - pos[d * 3 + 1];
  float dz = pos[s * 3 + 2] - pos[d * 3 + 2];
  dist[e] = sqrtf(dx * dx + dy * dy + dz * dz);
  atomicAdd(&deg[d], 1.0f);
}

// x = softplus(pos @ w_init + b_init)      [N,16]
__global__ void k_node_init(const float* __restrict__ pos,
                            const float* __restrict__ w,
                            const float* __restrict__ b,
                            float* __restrict__ x) {
  int t = blockIdx.x * blockDim.x + threadIdx.x;
  if (t >= N_NODES * 16) return;
  int n = t >> 4, c = t & 15;
  float v = b[c];
  v += pos[n * 3 + 0] * w[0 * 16 + c];
  v += pos[n * 3 + 1] * w[1 * 16 + c];
  v += pos[n * 3 + 2] * w[2 * 16 + c];
  x[t] = softplusf(v);
}

// xl = x@wl+bl, xr = x@wr+br               [N,32] each
__global__ void k_node_xform(const float* __restrict__ x,
                             const float* __restrict__ wl,
                             const float* __restrict__ bl,
                             const float* __restrict__ wr,
                             const float* __restrict__ br,
                             float* __restrict__ xl, float* __restrict__ xr) {
  int t = blockIdx.x * blockDim.x + threadIdx.x;
  if (t >= N_NODES * 32) return;
  int n = t >> 5, j = t & 31;
  float aL = bl[j], aR = br[j];
#pragma unroll
  for (int k = 0; k < 16; ++k) {
    float xv = x[n * 16 + k];
    aL = fmaf(xv, wl[k * 32 + j], aL);
    aR = fmaf(xv, wr[k * 32 + j], aR);
  }
  xl[t] = aL;
  xr[t] = aR;
}

// ---------------------------------------------------------------------------
// Fused: gaussian smearing (in-register) -> ep = ea @ we via WMMA f16 ->
// z = leakyrelu(xl[src]+xr[dst]+ep) -> logits = einsum(z, att) ->
// store logits, atomic segment-max per (dst, head).
// One wave32 handles a 16-edge tile; 4x v_wmma_f32_16x16x32_f16 per wave.
// B operands are pre-swizzled: one 32B contiguous load per fragment.
__global__ __launch_bounds__(256) void k_edge_logits_wmma(
    const float* __restrict__ dist, const int* __restrict__ ei,
    const v16h* __restrict__ bswz, const float* __restrict__ xl,
    const float* __restrict__ xr, const float* __restrict__ att,
    float* __restrict__ lex, unsigned* __restrict__ mmax) {
  const int lane = threadIdx.x & 31;
  const int wid = (blockIdx.x * blockDim.x + threadIdx.x) >> 5;
  const int e0 = wid * 16;
  const int c = lane & 15;  // column / row-in-tile for this lane
  const int hi = lane >> 4; // half-wave select (K split per ISA layout)

  // per-lane edge indices (lanes 16-31 mirror 0-15); rows fetched by shuffle
  const int eL = e0 + c;
  const int srcL = ei[eL];
  const int dstL = ei[N_EDGES + eL];

  // Build A (16x32 f16 per k-step) from SchNet smearing of dist.
  // ISA 16-bit A layout: lane m (0-15): halves 0..7 = K 0..7, 8..15 = K16..23;
  // lanes 16-31: K 8..15 / K24..31.
  const float d = dist[e0 + c];
  const float step = 5.0f / 49.0f;
  const float coeff = -0.5f / (step * step);
  v16h a0, a1;
#pragma unroll
  for (int j = 0; j < 16; ++j) {
    const int kb = hi * 8 + (j & 7) + ((j >> 3) << 4);
    {
      const float dd = d - step * (float)kb;
      a0[j] = (_Float16)__expf(coeff * dd * dd);
    }
    {
      const int k = kb + 32;
      float v = 0.0f;
      if (k < NGAUSS) {
        const float dd = d - step * (float)k;
        v = __expf(coeff * dd * dd);
      }
      a1[j] = (_Float16)v;
    }
  }

  // B fragments: contiguous per-lane 32B chunks (L0/L2 resident, 8KB table).
  v8f acc[2];
#pragma unroll
  for (int nt = 0; nt < 2; ++nt) {
    const v16h b0 = bswz[(nt * 2 + 0) * 32 + lane];
    const v16h b1 = bswz[(nt * 2 + 1) * 32 + lane];
    v8f a_ = {};
    a_ = __builtin_amdgcn_wmma_f32_16x16x32_f16(false, a0, false, b0,
                                                (short)0, a_, false, false);
    a_ = __builtin_amdgcn_wmma_f32_16x16x32_f16(false, a1, false, b1,
                                                (short)0, a_, false, false);
    acc[nt] = a_;
  }

  // Attention logits. D layout: acc[r] holds (edge m = r+8*hi, channel c).
#pragma unroll
  for (int h = 0; h < 2; ++h) {
    const float attv = att[h * 16 + c];
    const v8f aH = acc[h];
#pragma unroll
    for (int r = 0; r < 8; ++r) {
      const int m = r + 8 * hi;
      const int s_ = __shfl(srcL, m, 32);
      const int t_ = __shfl(dstL, m, 32);
      // 16 lanes of this half-wave read one contiguous 64B row chunk
      const float xlv = xl[s_ * 32 + h * 16 + c];
      const float xrv = xr[t_ * 32 + h * 16 + c];
      float z = xlv + xrv + aH[r];
      z = (z > 0.0f) ? z : 0.2f * z;  // leaky_relu slope 0.2
      float p = z * attv;
      p += __shfl_xor(p, 1, 32);
      p += __shfl_xor(p, 2, 32);
      p += __shfl_xor(p, 4, 32);
      p += __shfl_xor(p, 8, 32);
      if (c == r) {
        lex[(e0 + m) * 2 + h] = p;
        atomicMax(&mmax[t_ * 2 + h], fkey(p));
      }
    }
  }
}

// ex = exp(logit - max[dst]); segment-sum into ssum; overwrite lex with ex
__global__ void k_edge_exp(const int* __restrict__ ei,
                           const unsigned* __restrict__ mmax,
                           float* __restrict__ lex,
                           float* __restrict__ ssum) {
  int t = blockIdx.x * blockDim.x + threadIdx.x;
  if (t >= N_EDGES * 2) return;
  int e = t >> 1, h = t & 1;
  int d_ = ei[N_EDGES + e];
  float m = finv(mmax[d_ * 2 + h]);
  float ex = __expf(lex[t] - m);
  atomicAdd(&ssum[d_ * 2 + h], ex);
  lex[t] = ex;
}

// agg[dst, col] += xl[src, col] * alpha[e, head(col)]
__global__ void k_edge_aggr(const int* __restrict__ ei,
                            const float* __restrict__ lex,
                            const float* __restrict__ ssum,
                            const float* __restrict__ xl,
                            float* __restrict__ agg) {
  int t = blockIdx.x * blockDim.x + threadIdx.x;
  if (t >= N_EDGES * 32) return;
  int e = t >> 5, col = t & 31, h = col >> 4;
  int s_ = ei[e], d_ = ei[N_EDGES + e];
  float alpha = lex[e * 2 + h] / ssum[d_ * 2 + h];
  atomicAdd(&agg[d_ * 32 + col], xl[s_ * 32 + col] * alpha);
}

// out = softplus( 0.5*(agg_h0+agg_h1)/max(deg,1) + bias )
__global__ void k_node_final(const float* __restrict__ agg,
                             const float* __restrict__ deg,
                             const float* __restrict__ bias,
                             float* __restrict__ xout) {
  int t = blockIdx.x * blockDim.x + threadIdx.x;
  if (t >= N_NODES * 16) return;
  int n = t >> 4, c = t & 15;
  float a = 0.5f * (agg[n * 32 + c] + agg[n * 32 + 16 + c]);
  float v = a / fmaxf(deg[n], 1.0f) + bias[c];
  xout[t] = softplusf(v);
}

// y = softplus(x@wp1+bp1); scores = (y@wp2+bp2)/sigma
__global__ void k_head(const float* __restrict__ x,
                       const float* __restrict__ wp1,
                       const float* __restrict__ bp1,
                       const float* __restrict__ wp2,
                       const float* __restrict__ bp2,
                       const float* __restrict__ sig,
                       float* __restrict__ out) {
  int n = blockIdx.x * blockDim.x + threadIdx.x;
  if (n >= N_NODES) return;
  float xr[16], y[16];
#pragma unroll
  for (int k = 0; k < 16; ++k) xr[k] = x[n * 16 + k];
#pragma unroll
  for (int c = 0; c < 16; ++c) {
    float v = bp1[c];
#pragma unroll
    for (int k = 0; k < 16; ++k) v = fmaf(xr[k], wp1[k * 16 + c], v);
    y[c] = softplusf(v);
  }
  float inv_s = 1.0f / sig[n];
#pragma unroll
  for (int d = 0; d < 3; ++d) {
    float v = bp2[d];
#pragma unroll
    for (int k = 0; k < 16; ++k) v = fmaf(y[k], wp2[k * 3 + d], v);
    out[n * 3 + d] = v * inv_s;
  }
}

// ---------------------------------------------------------------------------
extern "C" void kernel_launch(void* const* d_in, const int* in_sizes, int n_in,
                              void* d_out, int out_size, void* d_ws,
                              size_t ws_size, hipStream_t stream) {
  const float* pos    = (const float*)d_in[0];
  const int*   ei     = (const int*)  d_in[1];
  const float* sig    = (const float*)d_in[2];
  const float* w_init = (const float*)d_in[3];
  const float* b_init = (const float*)d_in[4];
  const float* w_p1   = (const float*)d_in[5];
  const float* b_p1   = (const float*)d_in[6];
  const float* w_p2   = (const float*)d_in[7];
  const float* b_p2   = (const float*)d_in[8];
  // g1: 9..15 (wl,bl,wr,br,we,att,bias), g2: 16..22
  const float* g_wl[2]   = {(const float*)d_in[9],  (const float*)d_in[16]};
  const float* g_bl[2]   = {(const float*)d_in[10], (const float*)d_in[17]};
  const float* g_wr[2]   = {(const float*)d_in[11], (const float*)d_in[18]};
  const float* g_br[2]   = {(const float*)d_in[12], (const float*)d_in[19]};
  const float* g_we[2]   = {(const float*)d_in[13], (const float*)d_in[20]};
  const float* g_att[2]  = {(const float*)d_in[14], (const float*)d_in[21]};
  const float* g_bias[2] = {(const float*)d_in[15], (const float*)d_in[22]};

  // workspace carve-up (256B aligned)
  char* base = (char*)d_ws;
  size_t off = 0;
  auto carve = [&](size_t bytes) {
    char* p = base + off;
    off = (off + bytes + 255) & ~(size_t)255;
    return p;
  };
  float*    deg   = (float*)carve(N_NODES * 4);
  float*    xA    = (float*)carve(N_NODES * 16 * 4);
  float*    xB    = (float*)carve(N_NODES * 16 * 4);
  float*    xl    = (float*)carve(N_NODES * 32 * 4);
  float*    xr    = (float*)carve(N_NODES * 32 * 4);
  float*    dist  = (float*)carve(N_EDGES * 4);
  float*    lex   = (float*)carve(N_EDGES * 2 * 4);
  unsigned* mmax  = (unsigned*)carve(N_NODES * 2 * 4);
  float*    ssum  = (float*)carve(N_NODES * 2 * 4);
  float*    agg   = (float*)carve(N_NODES * 32 * 4);
  _Float16* bswz  = (_Float16*)carve(2 * 2 * 2 * 32 * 16 * 2);
  (void)n_in; (void)in_sizes; (void)out_size; (void)ws_size;

  const int B = 256;

  // setup
  hipMemsetAsync(deg, 0, N_NODES * 4, stream);
  k_prep_bswz<<<(2 * 2 * 2 * 32 * 16 + B - 1) / B, B, 0, stream>>>(
      g_we[0], g_we[1], bswz);
  k_dist_deg<<<N_EDGES / B, B, 0, stream>>>(pos, ei, dist, deg);
  k_node_init<<<N_NODES * 16 / B, B, 0, stream>>>(pos, w_init, b_init, xA);

  float* xin = xA;
  float* xout = xB;
  for (int L = 0; L < 3; ++L) {
    const int w = (L == 0) ? 0 : 1;  // layer1 uses g1; layers 2&3 reuse g2
    k_node_xform<<<N_NODES * 32 / B, B, 0, stream>>>(
        xin, g_wl[w], g_bl[w], g_wr[w], g_br[w], xl, xr);
    hipMemsetAsync(mmax, 0, N_NODES * 2 * 4, stream);
    hipMemsetAsync(ssum, 0, N_NODES * 2 * 4, stream);
    hipMemsetAsync(agg, 0, N_NODES * 32 * 4, stream);
    // one wave per 16-edge tile, 8 waves/block
    k_edge_logits_wmma<<<N_EDGES / 16 / 8, B, 0, stream>>>(
        dist, ei, (const v16h*)(bswz + (size_t)w * 2048), xl, xr, g_att[w],
        lex, mmax);
    k_edge_exp<<<N_EDGES * 2 / B, B, 0, stream>>>(ei, mmax, lex, ssum);
    k_edge_aggr<<<N_EDGES * 32 / B, B, 0, stream>>>(ei, lex, ssum, xl, agg);
    k_node_final<<<N_NODES * 16 / B, B, 0, stream>>>(agg, deg, g_bias[w],
                                                     xout);
    float* t = xin; xin = xout; xout = t;
  }

  k_head<<<(N_NODES + B - 1) / B, B, 0, stream>>>(xin, w_p1, b_p1, w_p2, b_p2,
                                                  sig, (float*)d_out);
}
